// EllipseAreaNetwork_28767690948632
// MI455X (gfx1250) — matrix operand order repaired and checked by reading, not compiled, not run on hardware
//
#include <hip/hip_runtime.h>
#include <hip/hip_bf16.h>

typedef __attribute__((ext_vector_type(16))) _Float16 v16h;
typedef __attribute__((ext_vector_type(8)))  _Float16 v8h;
typedef __attribute__((ext_vector_type(8)))  float    v8f;

#define KCLUST 8
#define HID 64
#define MAXIT 300
#define TOLSQ (1e-4f * 1e-4f)

// Monotonic uint encoding of float: enc(a) < enc(b) <=> a < b.
// Sentinel 0 (from memset) is below every finite encoding, so it doubles
// as the "no in-edge" marker (reference: where(has_in, agg, 0) then relu).
__device__ __forceinline__ unsigned encf(float x) {
    unsigned u = __float_as_uint(x);
    return (u & 0x80000000u) ? ~u : (u | 0x80000000u);
}

// ---------------- xs = relu(lin_similar(cat[x, x])) ----------------
__global__ void k_xs(const float* __restrict__ x, const float* __restrict__ w,
                     const float* __restrict__ b, float* __restrict__ xs, int n) {
    int i = blockIdx.x * blockDim.x + threadIdx.x;
    if (i >= n) return;
    float x0 = x[2 * i], x1 = x[2 * i + 1];
    float v = x0 * (w[0] + w[2]) + x1 * (w[1] + w[3]) + b[0];
    xs[i] = fmaxf(v, 0.0f);
}

// ---------------- out-degree over src ----------------
__global__ void k_deg(const int* __restrict__ ei, int* __restrict__ deg, int E) {
    int e = blockIdx.x * blockDim.x + threadIdx.x;
    if (e >= E) return;
    atomicAdd(&deg[ei[e]], 1);
}

// ---------------- EdgeConv: m1 (VALU) -> m2 = relu(m1) @ W2^T (WMMA) -> segment_max ----------------
__global__ void __launch_bounds__(256) k_edgeconv(
    const float* __restrict__ xs, const int* __restrict__ ei,
    const float* __restrict__ w1, const float* __restrict__ b1,
    const float* __restrict__ w2, const float* __restrict__ b2,
    unsigned* __restrict__ aggU, int E)
{
    __shared__ __align__(16) _Float16 As[128][HID + 8];   // relu(m1), f16, padded rows (144B)
    __shared__ __align__(16) _Float16 Bs[HID][HID + 8];   // Bs[n][k] = w2[n][k]  (B[k][n] of GEMM)
    __shared__ int   sDst[128];
    __shared__ float sW1[2 * HID];
    __shared__ float sB1[HID];
    __shared__ float sB2[HID];

    int tid = threadIdx.x;

    // Stage W2 as f16 (once per block) plus W1/b1/b2.
    for (int i = tid; i < HID * HID; i += 256) {
        Bs[i >> 6][i & 63] = (_Float16)w2[i];
    }
    if (tid < 2 * HID) sW1[tid] = w1[tid];
    if (tid < HID)     sB1[tid] = b1[tid];
    if (tid >= HID && tid < 2 * HID) sB2[tid - HID] = b2[tid - HID];
    __syncthreads();

    // Phase 1: two threads per edge, each computes 32 hidden channels of relu(m1).
    int e0 = blockIdx.x * 128;
    int le = tid >> 1, hp = tid & 1;
    int e = e0 + le;
    int hbase = hp * 32;
    if (e < E) {
        int s = ei[e], d = ei[E + e];
        float xi = xs[d], xj = xs[s];
        float dx = xj - xi;
        #pragma unroll
        for (int h = 0; h < 32; ++h) {
            int hh = hbase + h;
            float v = xi * sW1[2 * hh] + dx * sW1[2 * hh + 1] + sB1[hh];
            As[le][hh] = (_Float16)fmaxf(v, 0.0f);
        }
        if (hp == 0) sDst[le] = d;
    } else {
        #pragma unroll
        for (int h = 0; h < 32; ++h) As[le][hbase + h] = (_Float16)0.0f;
        if (hp == 0) sDst[le] = -1;
    }
    __syncthreads();

    // Phase 2: wave32 WMMA. Each wave owns a 16-edge tile; K=64 -> 2 chunks; N=64 -> 4 tiles.
    int wave = tid >> 5;
    int lane = tid & 31;
    int m    = lane & 15;
    int hi   = lane >> 4;
    int rA   = wave * 16 + m;          // A-matrix: lane holds row M = lane%16

    // A fragments per ISA layout: lanes<16: K = {0..7, 16..23}; lanes>=16: K = {8..15, 24..31} (+32*kc)
    v16h aFrag[2];
    #pragma unroll
    for (int kc = 0; kc < 2; ++kc) {
        int kb = kc * 32 + hi * 8;
        v8h lo  = *(const v8h*)&As[rA][kb];
        v8h hi8 = *(const v8h*)&As[rA][kb + 16];
        #pragma unroll
        for (int j = 0; j < 8; ++j) { aFrag[kc][j] = lo[j]; aFrag[kc][8 + j] = hi8[j]; }
    }

    // Hoist the 8 per-lane destination rows once (C: VGPR r holds row M = r + 8*hi).
    int dRow[8];
    #pragma unroll
    for (int r = 0; r < 8; ++r) dRow[r] = sDst[wave * 16 + hi * 8 + r];

    #pragma unroll
    for (int nt = 0; nt < 4; ++nt) {
        int n = nt * 16 + m;           // B/C: lane holds column N = lane%16
        v8f c = {};
        #pragma unroll
        for (int kc = 0; kc < 2; ++kc) {
            int kbB = kc * 32 + hi * 8;    // B: same K layout as A ({0..7,16..23} / {8..15,24..31})
            v8h b0 = *(const v8h*)&Bs[n][kbB];
            v8h b1v = *(const v8h*)&Bs[n][kbB + 16];
            v16h bFrag;
            #pragma unroll
            for (int j = 0; j < 8; ++j) { bFrag[j] = b0[j]; bFrag[8 + j] = b1v[j]; }
            c = __builtin_amdgcn_wmma_f32_16x16x32_f16(
                    false, aFrag[kc], false, bFrag, (short)0, c, false, false);
        }
        float bb = sB2[n];
        #pragma unroll
        for (int r = 0; r < 8; ++r) {
            int d = dRow[r];
            if (d >= 0) {
                float m2 = c[r] + bb;
                atomicMax(&aggU[(unsigned)d * 64u + (unsigned)n], encf(m2));
            }
        }
    }
}

// ---------------- xs2 = relu(where(has_in, agg, 0)) via decode ----------------
__global__ void k_xs2(const unsigned* __restrict__ aggU, float* __restrict__ xs2, int total) {
    int i = blockIdx.x * blockDim.x + threadIdx.x;
    if (i >= total) return;
    unsigned o = aggU[i];
    // top bit set => original value >= 0; otherwise negative or sentinel -> relu to 0
    xs2[i] = (o & 0x80000000u) ? __uint_as_float(o & 0x7fffffffu) : 0.0f;
}

// ---------------- k-means init: centroids from deterministic node indices; reset state ----------------
__global__ void k_kinit(const float* __restrict__ xs2, float* __restrict__ cents,
                        float* __restrict__ sums, float* __restrict__ cnts,
                        int* __restrict__ done, int n) {
    int i = threadIdx.x;                // 512 = 8 clusters * 64 dims
    if (i == 0) *done = 0;
    if (i < KCLUST) cnts[i] = 0.0f;
    int c = i >> 6, h = i & 63;
    unsigned idx = (2654435761u * (unsigned)c + 12345u) % (unsigned)n;
    cents[i] = xs2[(size_t)idx * 64u + (unsigned)h];
    sums[i] = 0.0f;
}

// ---------------- k-means assign + partial sums (conflict-free LDS reduce, then global atomics) ----------------
__global__ void __launch_bounds__(256) k_kassign(
    const float* __restrict__ xs2, const float* __restrict__ cents,
    int* __restrict__ cl, float* __restrict__ sums, float* __restrict__ cnts,
    const int* __restrict__ done, int n)
{
    if (*done) return;                  // uniform early-exit after convergence
    __shared__ float sc[KCLUST][HID];
    __shared__ float scn[KCLUST];
    __shared__ float ssum[KCLUST][HID];
    __shared__ float scnt[KCLUST];

    int tid = threadIdx.x;
    int lane = tid & 31;
    for (int i = tid; i < KCLUST * HID; i += 256) {
        sc[i >> 6][i & 63]   = cents[i];
        ssum[i >> 6][i & 63] = 0.0f;
    }
    if (tid < KCLUST) scnt[tid] = 0.0f;
    __syncthreads();
    if (tid < KCLUST) {
        float s = 0.0f;
        #pragma unroll
        for (int h = 0; h < HID; ++h) s += sc[tid][h] * sc[tid][h];
        scn[tid] = s;
    }
    __syncthreads();

    int nd = blockIdx.x * 256 + tid;
    if (nd < n) {
        float dot[KCLUST];
        #pragma unroll
        for (int c = 0; c < KCLUST; ++c) dot[c] = 0.0f;
        const float* xrow = xs2 + (size_t)nd * HID;
        for (int h = 0; h < HID; ++h) {
            float xv = xrow[h];
            #pragma unroll
            for (int c = 0; c < KCLUST; ++c) dot[c] += xv * sc[c][h];
        }
        // argmin of |x|^2 - 2 x.c + |c|^2 == argmin of |c|^2 - 2 x.c; first-min on ties
        int best = 0;
        float bv = scn[0] - 2.0f * dot[0];
        #pragma unroll
        for (int c = 1; c < KCLUST; ++c) {
            float dv = scn[c] - 2.0f * dot[c];
            if (dv < bv) { bv = dv; best = c; }
        }
        cl[nd] = best;
        atomicAdd(&scnt[best], 1.0f);
        // Lane-rotated h order: lanes sharing `best` hit 32 distinct LDS banks
        // each step instead of serializing on one address.
        for (int hh = 0; hh < HID; ++hh) {
            int h = (hh + lane) & (HID - 1);
            atomicAdd(&ssum[best][h], xrow[h]);
        }
    }
    __syncthreads();
    for (int i = tid; i < KCLUST * HID; i += 256) atomicAdd(&sums[i], ssum[i >> 6][i & 63]);
    if (tid < KCLUST) atomicAdd(&cnts[tid], scnt[tid]);
}

// ---------------- k-means centroid update + convergence flag; clear accumulators ----------------
__global__ void __launch_bounds__(512) k_kupdate(
    float* __restrict__ cents, float* __restrict__ sums,
    float* __restrict__ cnts, int* __restrict__ done)
{
    if (*done) return;
    __shared__ float red[512];
    int i = threadIdx.x;
    int c = i >> 6;
    float cnt = cnts[c];
    float oldv = cents[i];
    float nv = (cnt > 0.0f) ? (sums[i] / fmaxf(cnt, 1.0f)) : oldv;
    float df = nv - oldv;
    red[i] = df * df;
    __syncthreads();
    for (int s = 256; s > 0; s >>= 1) {
        if (i < s) red[i] += red[i + s];
        __syncthreads();
    }
    if (i == 0 && red[0] < TOLSQ) *done = 1;
    cents[i] = nv;
    sums[i] = 0.0f;
    if (i < KCLUST) cnts[i] = 0.0f;
}

// ---------------- per-edge polynomial + cluster-masked alpha combine + normalized scatter ----------------
__global__ void k_combined(const float* __restrict__ ea, const int* __restrict__ ei,
                           const int* __restrict__ cl, const int* __restrict__ deg,
                           const float* __restrict__ cw, const float* __restrict__ alpha,
                           float* __restrict__ outAcc, int E) {
    int e = blockIdx.x * blockDim.x + threadIdx.x;
    if (e >= E) return;
    int s = ei[e], d = ei[E + e];
    int ks = cl[s];
    if (ks != cl[d]) return;
    float a0 = ea[2 * e], a1 = ea[2 * e + 1];
    float h0 = a0 * cw[0] + a1 * cw[1];
    float h1 = a0 * cw[2] + a1 * cw[3];
    float h2 = a0 * cw[4] + a1 * cw[5];
    float t1 = (h1 > 0.0f) ? h1 : 0.1f * h1;          // leaky_relu^1
    float l2 = (h2 > 0.0f) ? h2 : 0.1f * h2;
    float t2 = l2 * l2;                                // leaky_relu^2
    const float* al = alpha + ks * 3;                  // alpha[k, i, 0, 0]
    float v = h0 * al[0] + t1 * al[1] + t2 * al[2];
    v /= (float)deg[s];
    atomicAdd(&outAcc[d], v);
}

// ---------------- global mean pool (sum of relu) ----------------
__global__ void __launch_bounds__(256) k_pool(const float* __restrict__ outAcc,
                                              float* __restrict__ total, int n) {
    __shared__ float red[256];
    int tid = threadIdx.x;
    float s = 0.0f;
    for (int i = blockIdx.x * 256 + tid; i < n; i += gridDim.x * 256)
        s += fmaxf(outAcc[i], 0.0f);
    red[tid] = s;
    __syncthreads();
    for (int st = 128; st > 0; st >>= 1) {
        if (tid < st) red[tid] += red[tid + st];
        __syncthreads();
    }
    if (tid == 0) atomicAdd(total, red[0]);
}

__global__ void k_final(const float* __restrict__ total, const float* __restrict__ fcw,
                        const float* __restrict__ fcb, float* __restrict__ out, float invn) {
    out[0] = (total[0] * invn) * fcw[0] + fcb[0];
}

extern "C" void kernel_launch(void* const* d_in, const int* in_sizes, int n_in,
                              void* d_out, int out_size, void* d_ws, size_t ws_size,
                              hipStream_t stream) {
    const float* x     = (const float*)d_in[0];
    const float* ea    = (const float*)d_in[1];
    const float* lsw   = (const float*)d_in[2];
    const float* lsb   = (const float*)d_in[3];
    const float* w1    = (const float*)d_in[4];
    const float* b1    = (const float*)d_in[5];
    const float* w2    = (const float*)d_in[6];
    const float* b2    = (const float*)d_in[7];
    const float* cw    = (const float*)d_in[8];
    const float* alpha = (const float*)d_in[9];
    const float* fcw   = (const float*)d_in[10];
    const float* fcb   = (const float*)d_in[11];
    const int*   ei    = (const int*)d_in[12];
    // d_in[13] = batch (all zeros, single graph) — handled implicitly.

    const int N = in_sizes[0] / 2;   // x is (N,2)
    const int E = in_sizes[1] / 2;   // edge_attr is (E,2)

    // Workspace layout (256B aligned chunks)
    char* ws = (char*)d_ws;
    size_t off = 0;
    auto alloc = [&](size_t bytes) -> size_t {
        size_t o = off;
        off = (off + bytes + 255) & ~(size_t)255;
        return o;
    };
    float*    xs     = (float*)(ws + alloc((size_t)N * 4));
    unsigned* aggU   = (unsigned*)(ws + alloc((size_t)N * HID * 4));
    float*    xs2    = (float*)(ws + alloc((size_t)N * HID * 4));
    int*      deg    = (int*)(ws + alloc((size_t)N * 4));
    int*      cl     = (int*)(ws + alloc((size_t)N * 4));
    float*    cents  = (float*)(ws + alloc(KCLUST * HID * 4));
    float*    sums   = (float*)(ws + alloc(KCLUST * HID * 4));
    float*    cnts   = (float*)(ws + alloc(KCLUST * 4));
    int*      done   = (int*)(ws + alloc(4));
    float*    total  = (float*)(ws + alloc(4));
    float*    outAcc = (float*)(ws + alloc((size_t)N * 4));
    (void)ws_size; (void)n_in; (void)out_size;

    // Reset all non-idempotent accumulators every call (graph replays don't re-poison).
    hipMemsetAsync(aggU, 0, (size_t)N * HID * 4, stream);   // 0 == sentinel below -inf encoding
    hipMemsetAsync(deg, 0, (size_t)N * 4, stream);
    hipMemsetAsync(outAcc, 0, (size_t)N * 4, stream);
    hipMemsetAsync(total, 0, 4, stream);

    const int nbN = (N + 255) / 256;
    k_xs<<<nbN, 256, 0, stream>>>(x, lsw, lsb, xs, N);
    k_deg<<<(E + 255) / 256, 256, 0, stream>>>(ei, deg, E);
    k_edgeconv<<<(E + 127) / 128, 256, 0, stream>>>(xs, ei, w1, b1, w2, b2, aggU, E);
    k_xs2<<<((N * HID) + 255) / 256, 256, 0, stream>>>(aggU, xs2, N * HID);

    k_kinit<<<1, 512, 0, stream>>>(xs2, cents, sums, cnts, done, N);
    for (int it = 0; it < MAXIT; ++it) {
        k_kassign<<<nbN, 256, 0, stream>>>(xs2, cents, cl, sums, cnts, done, N);
        k_kupdate<<<1, 512, 0, stream>>>(cents, sums, cnts, done);
    }

    k_combined<<<(E + 255) / 256, 256, 0, stream>>>(ea, ei, cl, deg, cw, alpha, outAcc, E);
    k_pool<<<256, 256, 0, stream>>>(outAcc, total, N);
    k_final<<<1, 1, 0, stream>>>(total, fcw, fcb, (float*)d_out, 1.0f / (float)N);
}
